// EQGATGNN_45552423141596
// MI455X (gfx1250) — compile-verified, matrix operand order checked by reading, not run on hardware
//
#include <hip/hip_runtime.h>
#include <hip/hip_bf16.h>
#include <math.h>

#define NN 30000
#define EE 480000
#define SI_ 64
#define VI_ 16

typedef __attribute__((ext_vector_type(16))) __bf16 v16bf;
typedef __attribute__((ext_vector_type(8))) __bf16 v8bf;
typedef __attribute__((ext_vector_type(8))) float v8f;

__device__ __forceinline__ v8f wmma_bf16f32(v16bf a, v16bf b, v8f c) {
  return __builtin_amdgcn_wmma_f32_16x16x32_bf16(false, a, false, b, (short)0, c,
                                                 false, false);
}

__device__ __forceinline__ v16bf combine16(v8bf lo, v8bf hi) {
  v16bf r;
#pragma unroll
  for (int e = 0; e < 8; ++e) {
    r[e] = lo[e];
    r[8 + e] = hi[e];
  }
  return r;
}

__device__ __forceinline__ v8bf cvt8(float4 a, float4 b) {
  v8bf r;
  r[0] = (__bf16)a.x; r[1] = (__bf16)a.y; r[2] = (__bf16)a.z; r[3] = (__bf16)a.w;
  r[4] = (__bf16)b.x; r[5] = (__bf16)b.y; r[6] = (__bf16)b.z; r[7] = (__bf16)b.w;
  return r;
}

__device__ __forceinline__ float4 add4(float4 a, float4 b) {
  return make_float4(a.x + b.x, a.y + b.y, a.z + b.z, a.w + b.w);
}

__device__ __forceinline__ float silu_f(float x) { return x / (1.0f + __expf(-x)); }

__device__ __forceinline__ unsigned fenc(float f) {
  int i = __float_as_int(f);
  return (i < 0) ? (unsigned)(~i) : (((unsigned)i) | 0x80000000u);
}
__device__ __forceinline__ float fdec(unsigned u) {
  return (u & 0x80000000u) ? __int_as_float((int)(u & 0x7fffffffu))
                           : __int_as_float(~(int)u);
}

// ---------------------------------------------------------------- utilities
__global__ void zero_kernel(float* p, int n) {
  int i = blockIdx.x * blockDim.x + threadIdx.x;
  if (i < n) p[i] = 0.0f;
}

__global__ void add_kernel(float* dstp, const float* srcp, int n) {
  int i = blockIdx.x * blockDim.x + threadIdx.x;
  if (i < n) dstp[i] += srcp[i];
}

__global__ void geom_kernel(const float* __restrict__ pos, const int* __restrict__ src,
                            const int* __restrict__ dst, float* __restrict__ d_e,
                            float* __restrict__ a_e, float* __restrict__ rn,
                            float* __restrict__ deg) {
  int e = blockIdx.x * blockDim.x + threadIdx.x;
  if (e >= EE) return;
  int se = src[e], de = dst[e];
  float px = pos[de * 3 + 0], py = pos[de * 3 + 1], pz = pos[de * 3 + 2];
  float qx = pos[se * 3 + 0], qy = pos[se * 3 + 1], qz = pos[se * 3 + 2];
  float rx = px - qx, ry = py - qy, rz = pz - qz;
  float a = px * qx + py * qy + pz * qz;
  float dd = sqrtf(fmaxf(rx * rx + ry * ry + rz * rz, 1e-6f));
  d_e[e] = dd;
  a_e[e] = a;
  rn[e * 3 + 0] = rx / dd;
  rn[e * 3 + 1] = ry / dd;
  rn[e * 3 + 2] = rz / dd;
  atomicAdd(&deg[de], 1.0f);
}

// Generic bf16-WMMA GEMM. K, N compile-time; M multiple of 16.
// W^T staged in LDS as bf16 [col][KPAD] zero-padded -> branch-free fragments.
// C stores: one base + constant-offset stores (N compile-time).
template <int K, int N, int ACT, bool HASBIAS>
__global__ __launch_bounds__(256) void gemm_bf16_kernel(
    const float* __restrict__ A, const float* __restrict__ W,
    const float* __restrict__ bias, float* __restrict__ C, int M) {
  constexpr int KPAD = ((K + 31) / 32) * 32;
  constexpr int KCC = KPAD / 32;
  constexpr int NTILES = N / 16;
  __shared__ __bf16 wt[N * KPAD];
  const int tid = threadIdx.x;
  // stage W^T (coalesced global reads; constant-divisor div)
  for (int i = tid; i < K * N; i += 256) {
    int k = i / N, col = i - k * N;
    wt[col * KPAD + k] = (__bf16)W[i];
  }
  if (KPAD > K) {
    constexpr int PADK = KPAD - K;
    for (int i = tid; i < N * PADK; i += 256) {
      int col = i / PADK, k = K + (i - col * PADK);
      wt[col * KPAD + k] = (__bf16)0.0f;
    }
  }
  __syncthreads();

  const int lane = tid & 31;
  const int wave = blockIdx.x * 8 + (tid >> 5);
  if (wave >= (M >> 4) * NTILES) return;
  const int mt = wave / NTILES;
  const int nt = wave - mt * NTILES;
  const int lm = lane & 15;
  const int half = lane >> 4;
  const int koff = half * 8;
  const float* arow = A + (mt * 16 + lm) * K;
  const __bf16* wcol = &wt[(nt * 16 + lm) * KPAD];

  v8f acc = {};
#pragma unroll
  for (int kc = 0; kc < KCC; ++kc) {
    const int kb = kc * 32;
    v8bf alo, ahi;
    if (kb + 15 < K) {
      float4 a0 = *(const float4*)(arow + kb + koff);
      float4 a1 = *(const float4*)(arow + kb + koff + 4);
      alo = cvt8(a0, a1);
    } else {
      alo = (v8bf)0;
    }
    if (kb + 31 < K) {
      float4 a2 = *(const float4*)(arow + kb + 16 + koff);
      float4 a3 = *(const float4*)(arow + kb + 16 + koff + 4);
      ahi = cvt8(a2, a3);
    } else {
      ahi = (v8bf)0;
    }
    v8bf blo = *(const v8bf*)(wcol + kb + koff);
    v8bf bhi = *(const v8bf*)(wcol + kb + 16 + koff);
    acc = wmma_bf16f32(combine16(alo, ahi), combine16(blo, bhi), acc);
  }

  float bb = 0.0f;
  if (HASBIAS) bb = bias[nt * 16 + lm];
  float* cp = C + (mt * 16 + half * 8) * N + nt * 16 + lm;
#pragma unroll
  for (int r = 0; r < 8; ++r) {
    float x = acc[r] + bb;
    if (ACT == 1) x = silu_f(x);
    cp[r * N] = x;  // constant byte offsets off one base
  }
}

// Edge MLP: one wave per 16-edge tile; weights transposed+padded in LDS.
template <int HASV>
__global__ __launch_bounds__(256) void edge_mlp_kernel(
    const float* __restrict__ s, const int* __restrict__ src,
    const int* __restrict__ dst, const float* __restrict__ d_e,
    const float* __restrict__ a_e, const float* __restrict__ W1,
    const float* __restrict__ b1, const float* __restrict__ W2,
    const float* __restrict__ b2, float* __restrict__ aij,
    float* __restrict__ vij0, float* __restrict__ vij1, float* __restrict__ vij2,
    float* __restrict__ gij) {
  constexpr int DOUT = HASV ? (3 * VI_ + SI_ + 1) : (VI_ + SI_ + 1);  // 113 / 81
  constexpr int NCT2 = HASV ? 8 : 6;                                  // padded /16
  __shared__ __bf16 w1t[64 * 64];        // W1[0:64]^T  (8KB)
  __shared__ __bf16 w2t[NCT2 * 16 * 64]; // W2^T zero-padded cols (16/12KB)
  __shared__ float wext[2 * 64];         // W1 rows 64(d),65(a)
  __shared__ float b1s[64];
  __shared__ float b2s[NCT2 * 16];
  __shared__ __bf16 h1b[8][16 * 64];     // bf16 h1 per wave (16KB)

  const int tid = threadIdx.x;
  for (int i = tid; i < 64 * 64; i += 256) {
    int k = i >> 6, col = i & 63;                 // global idx k*64+col == i
    w1t[col * 64 + k] = (__bf16)W1[i];
  }
  for (int i = tid; i < 64 * DOUT; i += 256) {
    int k = i / DOUT, col = i - k * DOUT;         // global idx == i (coalesced)
    w2t[col * 64 + k] = (__bf16)W2[i];
  }
  for (int i = tid; i < (NCT2 * 16 - DOUT) * 64; i += 256) {
    int col = DOUT + (i >> 6), k = i & 63;
    w2t[col * 64 + k] = (__bf16)0.0f;
  }
  if (tid < 64) {
    b1s[tid] = b1[tid];
    wext[tid] = W1[64 * 64 + tid];
    wext[64 + tid] = W1[65 * 64 + tid];
  }
  if (tid >= 128 && tid < 128 + NCT2 * 16) {
    int c = tid - 128;
    b2s[c] = (c < DOUT) ? b2[c] : 0.0f;
  }
  __syncthreads();

  const int wib = tid >> 5;
  const int lane = tid & 31;
  const int tile = blockIdx.x * 8 + wib;
  const int m = lane & 15;
  const int half = lane >> 4;
  const int koff = half * 8;
  const int eidx = tile * 16 + m;
  const int se = src[eidx], de = dst[eidx];
  const float myd = d_e[eidx], mya = a_e[eidx];
  const float* srow = s + se * SI_;
  const float* drow = s + de * SI_;

  // A fragments: two contiguous 8-float runs per 32-k chunk -> b128 loads
  v16bf afr[2];
#pragma unroll
  for (int kc = 0; kc < 2; ++kc) {
    const int kb = kc * 32;
    float4 s0 = add4(*(const float4*)(srow + kb + koff),
                     *(const float4*)(drow + kb + koff));
    float4 s1 = add4(*(const float4*)(srow + kb + koff + 4),
                     *(const float4*)(drow + kb + koff + 4));
    float4 s2 = add4(*(const float4*)(srow + kb + 16 + koff),
                     *(const float4*)(drow + kb + 16 + koff));
    float4 s3 = add4(*(const float4*)(srow + kb + 16 + koff + 4),
                     *(const float4*)(drow + kb + 16 + koff + 4));
    afr[kc] = combine16(cvt8(s0, s1), cvt8(s2, s3));
  }

  // GEMM1: h1 = silu(A @ W1[0:64] + d*W1[64] + a*W1[65] + b1)
#pragma unroll
  for (int ct = 0; ct < 4; ++ct) {
    const int col = ct * 16 + m;
    const __bf16* wc = &w1t[col * 64];
    v8f acc = {};
    acc = wmma_bf16f32(afr[0],
                       combine16(*(const v8bf*)(wc + koff),
                                 *(const v8bf*)(wc + 16 + koff)),
                       acc);
    acc = wmma_bf16f32(afr[1],
                       combine16(*(const v8bf*)(wc + 32 + koff),
                                 *(const v8bf*)(wc + 48 + koff)),
                       acc);
    const float bb = b1s[col];
    const float w64 = wext[col];
    const float w65 = wext[64 + col];
#pragma unroll
    for (int r = 0; r < 8; ++r) {
      int mr = r + half * 8;
      float dm = __shfl(myd, mr);
      float am = __shfl(mya, mr);
      float x = acc[r] + bb + dm * w64 + am * w65;
      h1b[wib][mr * 64 + col] = (__bf16)silu_f(x);
    }
  }
  asm volatile("s_wait_dscnt 0" ::: "memory");
  __builtin_amdgcn_wave_barrier();

  // re-fragment h1 (bf16 LDS, aligned b128 ds loads)
  v16bf hfr[2];
#pragma unroll
  for (int kc = 0; kc < 2; ++kc) {
    const __bf16* hr = &h1b[wib][m * 64 + kc * 32];
    hfr[kc] = combine16(*(const v8bf*)(hr + koff), *(const v8bf*)(hr + 16 + koff));
  }

  // GEMM2: h2 = h1 @ W2 + b2 ; route outputs
#pragma unroll
  for (int ct = 0; ct < NCT2; ++ct) {
    const int col = ct * 16 + m;
    const __bf16* wc = &w2t[col * 64];
    v8f acc = {};
    acc = wmma_bf16f32(hfr[0],
                       combine16(*(const v8bf*)(wc + koff),
                                 *(const v8bf*)(wc + 16 + koff)),
                       acc);
    acc = wmma_bf16f32(hfr[1],
                       combine16(*(const v8bf*)(wc + 32 + koff),
                                 *(const v8bf*)(wc + 48 + koff)),
                       acc);
    if (col < DOUT) {
      const float bb = b2s[col];
      const int e0 = tile * 16 + half * 8;
#pragma unroll
      for (int r = 0; r < 8; ++r) {
        int er = e0 + r;
        float x = acc[r] + bb;
        if (col < SI_) {
          aij[er * SI_ + col] = x;
        } else if (HASV) {
          if (col < 80) vij0[er * VI_ + (col - 64)] = x;
          else if (col < 96) vij1[er * VI_ + (col - 80)] = x;
          else if (col < 112) vij2[er * VI_ + (col - 96)] = x;
          else gij[er] = x;
        } else {
          if (col < 80) vij0[er * VI_ + (col - 64)] = x;
          else gij[er] = x;
        }
      }
    }
  }
}

// Scatter softmax pass 1: segment max via monotone-uint atomicMax.
__global__ void smax_kernel(const float* __restrict__ aij, const int* __restrict__ dst,
                            unsigned* __restrict__ nmax) {
  int i = blockIdx.x * blockDim.x + threadIdx.x;
  if (i >= EE * SI_) return;
  int e = i >> 6, c = i & 63;
  atomicMax(&nmax[dst[e] * SI_ + c], fenc(aij[i]));
}

// Pass 2: denom += exp(x-m); num += exp(x-m)*sb[src]
__global__ void sacc_kernel(const float* __restrict__ aij, const int* __restrict__ src,
                            const int* __restrict__ dst,
                            const unsigned* __restrict__ nmax,
                            const float* __restrict__ sbuf, float* __restrict__ denom,
                            float* __restrict__ nsnum) {
  int i = blockIdx.x * blockDim.x + threadIdx.x;
  if (i >= EE * SI_) return;
  int e = i >> 6, c = i & 63;
  int de = dst[e], se = src[e];
  float mm = fdec(nmax[de * SI_ + c]);
  if (!(mm == mm) || fabsf(mm) > 3.0e38f) mm = 0.0f;
  float ex = __expf(aij[i] - mm);
  atomicAdd(&denom[de * SI_ + c], ex);
  atomicAdd(&nsnum[de * SI_ + c], ex * sbuf[se * SI_ + c]);
}

__global__ void vmsg_kernel(const float* __restrict__ rn, const float* __restrict__ vij0,
                            const float* __restrict__ vij1,
                            const float* __restrict__ vij2,
                            const float* __restrict__ gij, const float* __restrict__ v,
                            const float* __restrict__ vb, const int* __restrict__ src,
                            const int* __restrict__ dst, float* __restrict__ mv,
                            float* __restrict__ mp, int hasv) {
  int i = blockIdx.x * blockDim.x + threadIdx.x;
  if (i >= EE * VI_) return;
  int e = i >> 4, c = i & 15;
  int se = src[e], de = dst[e];
  float r0 = rn[e * 3 + 0], r1 = rn[e * 3 + 1], r2 = rn[e * 3 + 2];
  float w0 = vij0[i];
  float n0 = r0 * w0, n1 = r1 * w0, n2 = r2 * w0;
  if (hasv) {
    float w1 = vij1[i], w2 = vij2[i];
    n0 += w1 * vb[se * 48 + c];
    n1 += w1 * vb[se * 48 + 16 + c];
    n2 += w1 * vb[se * 48 + 32 + c];
    float ad0 = v[de * 48 + c], ad1 = v[de * 48 + 16 + c], ad2 = v[de * 48 + 32 + c];
    float as0 = v[se * 48 + c], as1 = v[se * 48 + 16 + c], as2 = v[se * 48 + 32 + c];
    n0 += w2 * (ad1 * as2 - ad2 * as1);
    n1 += w2 * (ad2 * as0 - ad0 * as2);
    n2 += w2 * (ad0 * as1 - ad1 * as0);
  }
  atomicAdd(&mv[de * 48 + c], n0);
  atomicAdd(&mv[de * 48 + 16 + c], n1);
  atomicAdd(&mv[de * 48 + 32 + c], n2);
  if (c < 3) atomicAdd(&mp[de * 3 + c], gij[e] * rn[e * 3 + c]);
}

__global__ void nupd_kernel(float* __restrict__ s, float* __restrict__ v,
                            float* __restrict__ p, const float* __restrict__ nsnum,
                            const float* __restrict__ denom,
                            const float* __restrict__ mv, const float* __restrict__ mp,
                            const float* __restrict__ deg) {
  int i = blockIdx.x * blockDim.x + threadIdx.x;
  if (i >= NN * SI_) return;
  int n = i >> 6, j = i & 63;
  float dg = fmaxf(deg[n], 1.0f);
  s[i] += nsnum[i] / (denom[i] + 1e-16f);
  if (j < 48) v[n * 48 + j] += mv[n * 48 + j] / dg;
  if (j < 3) p[n * 3 + j] += mp[n * 3 + j] / dg;
}

__global__ void shbuild_kernel(const float* __restrict__ s, const float* __restrict__ vv,
                               float* __restrict__ sh) {
  int i = blockIdx.x * blockDim.x + threadIdx.x;
  if (i >= NN * 80) return;
  int n = i / 80, j = i - n * 80;
  if (j < SI_) {
    sh[i] = s[n * SI_ + j];
  } else {
    int c = j - SI_;
    float x0 = vv[(n * 3 + 0) * 32 + c];
    float x1 = vv[(n * 3 + 1) * 32 + c];
    float x2 = vv[(n * 3 + 2) * 32 + c];
    sh[i] = sqrtf(fmaxf(x0 * x0 + x1 * x1 + x2 * x2, 1e-6f));
  }
}

__global__ void gate_kernel(const float* __restrict__ sh2, const float* __restrict__ vv,
                            float* __restrict__ s, float* __restrict__ vtmp) {
  int i = blockIdx.x * blockDim.x + threadIdx.x;
  if (i >= NN * 80) return;
  int n = i / 80, j = i - n * 80;
  if (j < 16) {
    float g = sh2[i];
#pragma unroll
    for (int d = 0; d < 3; ++d)
      vtmp[(n * 3 + d) * VI_ + j] = g * vv[(n * 3 + d) * 32 + VI_ + j];
  } else {
    s[n * SI_ + (j - 16)] += sh2[i];
  }
}

// -------------------------------------------------------------------- host
template <int K, int N, int ACT, bool HASBIAS>
static void launch_gemm(const float* A, const float* W, const float* bias, float* C,
                        int M, hipStream_t st) {
  int waves = (M / 16) * (N / 16);
  gemm_bf16_kernel<K, N, ACT, HASBIAS><<<(waves + 7) / 8, 256, 0, st>>>(A, W, bias,
                                                                        C, M);
}

extern "C" void kernel_launch(void* const* d_in, const int* in_sizes, int n_in,
                              void* d_out, int out_size, void* d_ws, size_t ws_size,
                              hipStream_t stream) {
  (void)in_sizes; (void)n_in; (void)out_size; (void)ws_size;
  const float* in_s = (const float*)d_in[0];
  const float* in_v = (const float*)d_in[1];
  const float* in_p = (const float*)d_in[2];
  const float* pos = (const float*)d_in[3];
  const int* eix = (const int*)d_in[4];
  const int* src = eix;
  const int* dst = eix + EE;

  struct Layer {
    const float *ew1, *eb1, *ew2, *eb2, *sw, *sb, *wv0, *vecw;
    const float *ws1, *ws1b, *ws2, *ws2b, *wv1, *ws, *wsb;
    int hasv, mlp;
  } L[5];
  int ci = 5;
  for (int i = 0; i < 5; ++i) {
    L[i].hasv = (i > 0);
    L[i].mlp = (i < 4);
    L[i].ew1 = (const float*)d_in[ci++];
    L[i].eb1 = (const float*)d_in[ci++];
    L[i].ew2 = (const float*)d_in[ci++];
    L[i].eb2 = (const float*)d_in[ci++];
    L[i].sw = (const float*)d_in[ci++];
    L[i].sb = (const float*)d_in[ci++];
    L[i].wv0 = (const float*)d_in[ci++];
    L[i].vecw = L[i].ws1 = L[i].ws1b = L[i].ws2 = L[i].ws2b = nullptr;
    L[i].wv1 = L[i].ws = L[i].wsb = nullptr;
    if (L[i].hasv) L[i].vecw = (const float*)d_in[ci++];
    if (L[i].mlp) {
      L[i].ws1 = (const float*)d_in[ci++];
      L[i].ws1b = (const float*)d_in[ci++];
      L[i].ws2 = (const float*)d_in[ci++];
      L[i].ws2b = (const float*)d_in[ci++];
      L[i].wv1 = (const float*)d_in[ci++];
    } else {
      L[i].ws = (const float*)d_in[ci++];
      L[i].wsb = (const float*)d_in[ci++];
    }
  }

  char* wp = (char*)d_ws;
  auto carve = [&](size_t nf) -> float* {
    float* r = (float*)wp;
    wp += ((nf * sizeof(float) + 255) / 256) * 256;
    return r;
  };
  float* ge_d = carve(EE);
  float* ge_a = carve(EE);
  float* rn = carve((size_t)EE * 3);
  float* deg = carve(NN);
  float* sb = carve((size_t)NN * SI_);
  float* vb = carve((size_t)NN * 48);
  float* aij = carve((size_t)EE * SI_);
  float* vij0 = carve((size_t)EE * VI_);
  float* vij1 = carve((size_t)EE * VI_);
  float* vij2 = carve((size_t)EE * VI_);
  float* gij = carve(EE);
  float* atom = carve((size_t)NN * 243);  // nmax|denom|nsnum|mv|mp contiguous
  unsigned* nmax = (unsigned*)atom;
  float* denom = atom + (size_t)NN * 64;
  float* nsnum = denom + (size_t)NN * 64;
  float* mv = nsnum + (size_t)NN * 64;
  float* mp = mv + (size_t)NN * 48;
  float* vv = carve((size_t)NN * 96);
  float* sh = carve((size_t)NN * 80);
  float* t1 = carve((size_t)NN * 64);
  float* sh2 = carve((size_t)NN * 80);
  float* vtmp = carve((size_t)NN * 48);
  float* vd = carve((size_t)NN * 48);

  float* s_st = (float*)d_out;
  float* v_st = s_st + (size_t)NN * SI_;
  float* p_st = v_st + (size_t)NN * 48;

  auto z = [&](float* p, size_t n) {
    zero_kernel<<<(unsigned)((n + 255) / 256), 256, 0, stream>>>(p, (int)n);
  };

  z(deg, NN);
  geom_kernel<<<(EE + 255) / 256, 256, 0, stream>>>(pos, src, dst, ge_d, ge_a, rn, deg);
  hipMemcpyAsync(s_st, in_s, (size_t)NN * SI_ * 4, hipMemcpyDeviceToDevice, stream);
  hipMemcpyAsync(v_st, in_v, (size_t)NN * 48 * 4, hipMemcpyDeviceToDevice, stream);
  hipMemcpyAsync(p_st, in_p, (size_t)NN * 3 * 4, hipMemcpyDeviceToDevice, stream);

  for (int i = 0; i < 5; ++i) {
    launch_gemm<64, 64, 0, true>(s_st, L[i].sw, L[i].sb, sb, NN, stream);
    if (L[i].hasv)
      launch_gemm<16, 16, 0, false>(v_st, L[i].vecw, nullptr, vb, NN * 3, stream);

    if (L[i].hasv)
      edge_mlp_kernel<1><<<EE / 128, 256, 0, stream>>>(
          s_st, src, dst, ge_d, ge_a, L[i].ew1, L[i].eb1, L[i].ew2, L[i].eb2, aij,
          vij0, vij1, vij2, gij);
    else
      edge_mlp_kernel<0><<<EE / 128, 256, 0, stream>>>(
          s_st, src, dst, ge_d, ge_a, L[i].ew1, L[i].eb1, L[i].ew2, L[i].eb2, aij,
          vij0, vij1, vij2, gij);

    z(atom, (size_t)NN * 243);
    smax_kernel<<<(EE * SI_ + 255) / 256, 256, 0, stream>>>(aij, dst, nmax);
    sacc_kernel<<<(EE * SI_ + 255) / 256, 256, 0, stream>>>(aij, src, dst, nmax, sb,
                                                            denom, nsnum);
    vmsg_kernel<<<(EE * VI_ + 255) / 256, 256, 0, stream>>>(
        rn, vij0, vij1, vij2, gij, v_st, vb, src, dst, mv, mp, L[i].hasv);
    nupd_kernel<<<(NN * SI_ + 255) / 256, 256, 0, stream>>>(s_st, v_st, p_st, nsnum,
                                                            denom, mv, mp, deg);
    // gated equivariant block
    launch_gemm<16, 32, 0, false>(v_st, L[i].wv0, nullptr, vv, NN * 3, stream);
    shbuild_kernel<<<(NN * 80 + 255) / 256, 256, 0, stream>>>(s_st, vv, sh);
    if (L[i].mlp) {
      launch_gemm<80, 64, 1, true>(sh, L[i].ws1, L[i].ws1b, t1, NN, stream);
      launch_gemm<64, 80, 0, true>(t1, L[i].ws2, L[i].ws2b, sh2, NN, stream);
    } else {
      launch_gemm<80, 80, 0, true>(sh, L[i].ws, L[i].wsb, sh2, NN, stream);
    }
    gate_kernel<<<(NN * 80 + 255) / 256, 256, 0, stream>>>(sh2, vv, s_st, vtmp);
    if (L[i].mlp) {
      launch_gemm<16, 16, 0, false>(vtmp, L[i].wv1, nullptr, vd, NN * 3, stream);
      add_kernel<<<(NN * 48 + 255) / 256, 256, 0, stream>>>(v_st, vd, NN * 48);
    } else {
      add_kernel<<<(NN * 48 + 255) / 256, 256, 0, stream>>>(v_st, vtmp, NN * 48);
    }
  }
}
